// DensePoseFCHead_50027779063884
// MI455X (gfx1250) — matrix operand order repaired and checked by reading, not compiled, not run on hardware
//
#include <hip/hip_runtime.h>

typedef __attribute__((ext_vector_type(16))) __bf16 v16bf;
typedef __attribute__((ext_vector_type(8)))  float  v8f;

#define HWDIM 28
#define HW2   784
#define NIMG  128
#define OCH   256

__constant__ int c_dh[9] = {-1, -1, -1, 0, 0, -2, 0, 2, 3};
__constant__ int c_dw[9] = {-3, -1, 1, 0, 2, 1, -1, 0, 1};

// plain POD 128-bit chunk (avoid HIP uint4 ctor issues in bit_cast)
struct alignas(16) Q4 { unsigned x, y, z, w; };
struct alignas(32) Q8 { Q4 lo, hi; };

__device__ __forceinline__ v16bf mk_frag(Q4 lo, Q4 hi) {
  Q8 p{lo, hi};
  return __builtin_bit_cast(v16bf, p);
}

__device__ __forceinline__ unsigned short f2bf(float f) {
  unsigned u = __builtin_bit_cast(unsigned, f);
  u += 0x7FFFu + ((u >> 16) & 1u);          // round-to-nearest-even
  return (unsigned short)(u >> 16);
}

// ---------------------------------------------------------------------------
// Prep: maxpool(fine_segm,4) + concat with features + f32->bf16, pad ch->Cpad
// out[n][c][hw] bf16
// ---------------------------------------------------------------------------
__global__ void prep_input(const float* __restrict__ feat,
                           const float* __restrict__ seg,
                           unsigned short* __restrict__ out, int Cpad) {
  long idx = (long)blockIdx.x * blockDim.x + threadIdx.x;
  long total = (long)NIMG * Cpad * HW2;
  if (idx >= total) return;
  int hw = (int)(idx % HW2);
  long t = idx / HW2;
  int c = (int)(t % Cpad);
  int n = (int)(t / Cpad);
  float v = 0.f;
  if (c < 512) {
    v = feat[((long)n * 512 + c) * HW2 + hw];
  } else if (c < 537) {
    int cc = c - 512, h = hw / HWDIM, w = hw % HWDIM;
    const float* p = seg + (((long)n * 25 + cc) * 112 + h * 4) * 112 + (long)w * 4;
    float m = p[0];
#pragma unroll
    for (int i = 0; i < 4; ++i)
#pragma unroll
      for (int j = 0; j < 4; ++j) { float z = p[i * 112 + j]; m = z > m ? z : m; }
    v = m;
  }
  out[idx] = f2bf(v);
}

// ---------------------------------------------------------------------------
// Weight repack: [O][C][9] f32 -> [O][tap*Cpad + c] bf16 (zero pad c>=Cin)
// ---------------------------------------------------------------------------
__global__ void repack_w(const float* __restrict__ w,
                         unsigned short* __restrict__ out,
                         int Cin, int Cpad) {
  long idx = (long)blockIdx.x * blockDim.x + threadIdx.x;
  long total = (long)OCH * 9 * Cpad;
  if (idx >= total) return;
  int c = (int)(idx % Cpad);
  long t2 = idx / Cpad;
  int t = (int)(t2 % 9);
  int o = (int)(t2 / 9);
  float v = (c < Cin) ? w[((long)o * Cin + c) * 9 + t] : 0.f;
  out[((long)o * 9 + t) * Cpad + c] = f2bf(v);
}

// ---------------------------------------------------------------------------
// Deform-conv (9 shifted GEMMs) via v_wmma_f32_16x16x32_bf16.
// Grid: (49 spatial tiles, 1, 128 images). Block: 256 threads = 8 waves.
// Wave w computes out-channel tiles [w*16,+16) and [128+w*16,+16).
// Workgroup stages 64K x 16N bf16 activation tile in double-buffered LDS;
// one barrier per stage, 4 WMMAs per wave per stage, prefetch of next stage.
// Cin must be a multiple of 64.
// ---------------------------------------------------------------------------
__global__ void __launch_bounds__(256)
dconv_wmma(const unsigned short* __restrict__ xin, int Cin,
           const unsigned short* __restrict__ wrep,
           unsigned short* __restrict__ out_bf,
           float* __restrict__ out_f32) {
  __shared__ __align__(32) unsigned short Bt[2][16][64];  // [buf][N][K], row=128B

  const int tid  = threadIdx.x;
  const int wave = tid >> 5;
  const int lane = tid & 31;
  const int m    = lane & 15;     // A row / B col / D col
  const int half = lane >> 4;
  const int n    = blockIdx.z;
  const int hw0  = blockIdx.x * 16;
  const int Ktot = 9 * Cin;

  v8f acc0 = {}; v8f acc1 = {};

  // staging mapping: thread -> (column j, 4-channel group)
  const int bj  = tid & 15;
  const int bkq = tid >> 4;            // 0..15 -> channels bkq*4 .. bkq*4+3
  const int hwj = hw0 + bj;
  const int hj  = hwj / HWDIM;
  const int wj  = hwj % HWDIM;

  const unsigned short* wbase0 = wrep + (size_t)(wave * 16 + m) * Ktot;
  const unsigned short* wbase1 = wrep + (size_t)(128 + wave * 16 + m) * Ktot;

  int stage = 0;
  for (int t = 0; t < 9; ++t) {
    const int dh = c_dh[t], dw = c_dw[t];
    const int h = hj + dh, w = wj + dw;
    const bool inb = (h >= 0) && (h < HWDIM) && (w >= 0) && (w < HWDIM);
    const unsigned short* src =
        xin + (long)n * Cin * HW2 + (long)(bkq * 4) * HW2 + (long)h * HWDIM + w;

    for (int c0 = 0; c0 < Cin; c0 += 64) {
      const int buf = stage & 1;
      unsigned p0 = 0, p1 = 0;
      if (inb) {
        const unsigned short* p = src + (long)c0 * HW2;
        p0 = (unsigned)p[0]       | ((unsigned)p[HW2]     << 16);
        p1 = (unsigned)p[2 * HW2] | ((unsigned)p[3 * HW2] << 16);
        if (c0 + 64 < Cin)
          __builtin_prefetch((const void*)(p + (long)64 * HW2), 0, 1);
      }
      *(unsigned long long*)&Bt[buf][bj][bkq * 4] =
          ((unsigned long long)p1 << 32) | p0;
      __syncthreads();   // single barrier/stage: prev stage's barrier protects WAR

#pragma unroll
      for (int sub = 0; sub < 2; ++sub) {
        // A fragments: K = kb..kb+7 and kb+16..kb+23 (ISA 16-bit A 16x32 layout)
        const int kb = t * Cin + c0 + sub * 32 + half * 8;
        Q4 a0 = *(const Q4*)(wbase0 + kb);
        Q4 a1 = *(const Q4*)(wbase0 + kb + 16);
        Q4 a2 = *(const Q4*)(wbase1 + kb);
        Q4 a3 = *(const Q4*)(wbase1 + kb + 16);
        // B fragment: K = kb..kb+7 and kb+16..kb+23 (matches A layout)
        Q4 b0 = *(const Q4*)&Bt[buf][m][sub * 32 + half * 8];
        Q4 b1 = *(const Q4*)&Bt[buf][m][sub * 32 + half * 8 + 16];
        v16bf B = mk_frag(b0, b1);

        acc0 = __builtin_amdgcn_wmma_f32_16x16x32_bf16(
            false, mk_frag(a0, a1), false, B, (short)0, acc0, false, false);
        acc1 = __builtin_amdgcn_wmma_f32_16x16x32_bf16(
            false, mk_frag(a2, a3), false, B, (short)0, acc1, false, false);
      }
      ++stage;
    }
  }

  // Epilogue: D layout -> VGPR r holds M = half*8 + r, column N = m
  const int hw = hw0 + m;
#pragma unroll
  for (int r = 0; r < 8; ++r) {
    int o0 = wave * 16 + half * 8 + r;
    float v0 = acc0[r]; v0 = v0 > 0.f ? v0 : 0.f;
    float v1 = acc1[r]; v1 = v1 > 0.f ? v1 : 0.f;
    size_t i0 = ((size_t)n * OCH + o0) * HW2 + hw;
    size_t i1 = ((size_t)n * OCH + o0 + 128) * HW2 + hw;
    if (out_bf)  { out_bf[i0] = f2bf(v0); out_bf[i1] = f2bf(v1); }
    if (out_f32) { out_f32[i0] = v0;      out_f32[i1] = v1; }
  }
}

// ---------------------------------------------------------------------------
extern "C" void kernel_launch(void* const* d_in, const int* in_sizes, int n_in,
                              void* d_out, int out_size, void* d_ws, size_t ws_size,
                              hipStream_t stream) {
  (void)in_sizes; (void)n_in; (void)out_size; (void)ws_size;
  const float* feat = (const float*)d_in[0];
  const float* seg  = (const float*)d_in[1];
  const float* w1   = (const float*)d_in[2];
  const float* w2   = (const float*)d_in[3];
  const float* w3   = (const float*)d_in[4];
  const float* w4   = (const float*)d_in[5];

  char* ws = (char*)d_ws;
  size_t off = 0;
  auto alloc = [&](size_t bytes) -> void* {
    void* p = ws + off;
    off = (off + bytes + 255) & ~(size_t)255;
    return p;
  };

  const int C1 = 576;                                   // 537 padded to /64
  unsigned short* xin0 = (unsigned short*)alloc((size_t)NIMG * C1 * HW2 * 2);
  unsigned short* wr1  = (unsigned short*)alloc((size_t)OCH * 9 * C1 * 2);
  unsigned short* wr2  = (unsigned short*)alloc((size_t)OCH * 9 * 256 * 2);
  unsigned short* wr3  = (unsigned short*)alloc((size_t)OCH * 9 * 256 * 2);
  unsigned short* wr4  = (unsigned short*)alloc((size_t)OCH * 9 * 256 * 2);
  unsigned short* x1   = (unsigned short*)alloc((size_t)NIMG * 256 * HW2 * 2);
  unsigned short* x2   = (unsigned short*)alloc((size_t)NIMG * 256 * HW2 * 2);
  unsigned short* x3   = xin0;                          // reuse (no longer needed)

  {
    long total = (long)NIMG * C1 * HW2;
    prep_input<<<dim3((unsigned)((total + 255) / 256)), 256, 0, stream>>>(feat, seg, xin0, C1);
  }
  {
    long t1 = (long)OCH * 9 * C1;
    repack_w<<<dim3((unsigned)((t1 + 255) / 256)), 256, 0, stream>>>(w1, wr1, 537, C1);
    long t2 = (long)OCH * 9 * 256;
    repack_w<<<dim3((unsigned)((t2 + 255) / 256)), 256, 0, stream>>>(w2, wr2, 256, 256);
    repack_w<<<dim3((unsigned)((t2 + 255) / 256)), 256, 0, stream>>>(w3, wr3, 256, 256);
    repack_w<<<dim3((unsigned)((t2 + 255) / 256)), 256, 0, stream>>>(w4, wr4, 256, 256);
  }

  dim3 grid(49, 1, NIMG), block(256);
  dconv_wmma<<<grid, block, 0, stream>>>(xin0, C1,  wr1, x1, nullptr);
  dconv_wmma<<<grid, block, 0, stream>>>(x1,   256, wr2, x2, nullptr);
  dconv_wmma<<<grid, block, 0, stream>>>(x2,   256, wr3, x3, nullptr);
  dconv_wmma<<<grid, block, 0, stream>>>(x3,   256, wr4, nullptr, (float*)d_out);
}
